// SwinBlock_66408784331146
// MI455X (gfx1250) — matrix operand order repaired and checked by reading, not compiled, not run on hardware
//
#include <hip/hip_runtime.h>
#include <hip/hip_bf16.h>
#include <cstdint>
#include <cstddef>

typedef __attribute__((ext_vector_type(16))) _Float16 v16h;
typedef __attribute__((ext_vector_type(8)))  _Float16 v8h;
typedef __attribute__((ext_vector_type(8)))  float    v8f;

#define DEV __device__ __forceinline__

// ---------------- problem constants ----------------
constexpr int  C_DIM   = 256;
constexpr int  N_HEAD  = 8;
constexpr int  D_HEAD  = 32;
constexpr int  N_TOK   = 49;                 // tokens per window
constexpr int  TOK     = 32 * 56 * 56;       // 100352 total tokens
constexpr int  NWIN    = TOK / N_TOK;        // 2048 windows (global)
constexpr int  MTILES  = TOK / 16;           // 6272

// ---------------- workspace layout (bytes) ----------------
constexpr size_t SZ_ACT16 = (size_t)TOK * C_DIM * 2;            // 51,380,224
constexpr size_t SZ_QK16  = (size_t)NWIN * N_HEAD * N_TOK * D_HEAD * 2; // 51,380,224
constexpr size_t SZ_VT16  = (size_t)NWIN * N_HEAD * D_HEAD * 64 * 2;    // 67,108,864 (K padded to 64)
constexpr size_t OFF_XR   = 0;                       // LN1+roll out (f16) ; reused as LN2 out
constexpr size_t OFF_Q    = OFF_XR + SZ_ACT16;
constexpr size_t OFF_K    = OFF_Q  + SZ_QK16;        // Q region reused as fc1-hidden
constexpr size_t OFF_VT   = OFF_K  + SZ_QK16;
constexpr size_t OFF_O    = OFF_VT + SZ_VT16;        // attention output (f16, window order)
constexpr size_t OFF_X1   = OFF_O  + SZ_ACT16;       // post-attn residual (f32)
constexpr size_t OFF_WQKV = OFF_X1 + (size_t)TOK * C_DIM * 4;
constexpr size_t OFF_WPRJ = OFF_WQKV + (size_t)768 * 256 * 2;
constexpr size_t OFF_WFC1 = OFF_WPRJ + (size_t)256 * 256 * 2;
constexpr size_t OFF_WFC2 = OFF_WFC1 + (size_t)256 * 256 * 2;

// ---------------- WMMA fragment loaders (contiguous k order, same for A and B) ----------------
// A (16x32, MxK): lane<16 -> row M=lane, K kk..kk+15;
//                 lane>=16 -> row M=lane-16, K kk+16..kk+31
DEV v16h ld_afrag(const _Float16* A, size_t row, int kk, int lane) {
    return *(const v16h*)(A + row * C_DIM + kk + ((lane & 16) ? 16 : 0));
}
// B (32x16, KxN) from row-major W[n][k]: lane<16 -> N=n, K kk..kk+15;
//                                        lane>=16 -> N=n, K kk+16..kk+31
DEV v16h ld_bfrag(const _Float16* W, size_t n, int kk, int lane) {
    return *(const v16h*)(W + n * C_DIM + kk + ((lane & 16) ? 16 : 0));
}

// wave computes a 16(M) x 64(N) tile of A[M,256] @ W[N,256]^T with f32 accum
DEV void gemm16x64(const _Float16* A, const _Float16* W,
                   int m0, int n0, int lane, v8f c[4]) {
    const int lr = lane & 15;
#pragma unroll
    for (int kk = 0; kk < C_DIM; kk += 32) {
        v16h a = ld_afrag(A, (size_t)(m0 + lr), kk, lane);
#pragma unroll
        for (int j = 0; j < 4; ++j) {
            v16h b = ld_bfrag(W, (size_t)(n0 + j * 16 + lr), kk, lane);
            c[j] = __builtin_amdgcn_wmma_f32_16x16x32_f16(
                false, a, false, b, (short)0, c[j], false, false);
        }
    }
}

// ---------------- small utility kernels ----------------
__global__ void k_zero32(uint32_t* p, size_t n32) {
    size_t i  = (size_t)blockIdx.x * blockDim.x + threadIdx.x;
    size_t st = (size_t)gridDim.x * blockDim.x;
    for (; i < n32; i += st) p[i] = 0u;
}

__global__ void k_cvt16(const float* s, _Float16* d, int n) {
    int i = blockIdx.x * blockDim.x + threadIdx.x;
    if (i < n) d[i] = (_Float16)s[i];
}

// ---------------- LN1 + roll(-3,-3) + window-order gather, f16 out ----------------
__global__ __launch_bounds__(256) void k_ln1(const float* __restrict__ x,
                                             const float* __restrict__ w,
                                             const float* __restrict__ b,
                                             _Float16* __restrict__ xr) {
    __shared__ float red[256];
    const int tid = threadIdx.x;
    const int t = blockIdx.x;                 // window-order token
    const int bi = t / 3136, p = t - bi * 3136;
    const int r = p / 56, c = p - r * 56;
    int sr = r + 3; if (sr >= 56) sr -= 56;
    int sc = c + 3; if (sc >= 56) sc -= 56;
    const size_t src = (((size_t)bi * 3136) + sr * 56 + sc) * C_DIM + tid;
    float v = x[src];
    red[tid] = v; __syncthreads();
#pragma unroll
    for (int s = 128; s > 0; s >>= 1) { if (tid < s) red[tid] += red[tid + s]; __syncthreads(); }
    const float mean = red[0] * (1.0f / 256.0f);
    __syncthreads();
    float d = v - mean;
    red[tid] = d * d; __syncthreads();
#pragma unroll
    for (int s = 128; s > 0; s >>= 1) { if (tid < s) red[tid] += red[tid + s]; __syncthreads(); }
    const float var = red[0] * (1.0f / 256.0f);
    xr[(size_t)t * C_DIM + tid] = (_Float16)(d * rsqrtf(var + 1e-5f) * w[tid] + b[tid]);
}

// ---------------- LN2 (natural token order) ----------------
__global__ __launch_bounds__(256) void k_ln2(const float* __restrict__ x1,
                                             const float* __restrict__ w,
                                             const float* __restrict__ b,
                                             _Float16* __restrict__ y2) {
    __shared__ float red[256];
    const int tid = threadIdx.x;
    const size_t base = (size_t)blockIdx.x * C_DIM;
    float v = x1[base + tid];
    red[tid] = v; __syncthreads();
#pragma unroll
    for (int s = 128; s > 0; s >>= 1) { if (tid < s) red[tid] += red[tid + s]; __syncthreads(); }
    const float mean = red[0] * (1.0f / 256.0f);
    __syncthreads();
    float d = v - mean;
    red[tid] = d * d; __syncthreads();
#pragma unroll
    for (int s = 128; s > 0; s >>= 1) { if (tid < s) red[tid] += red[tid + s]; __syncthreads(); }
    const float var = red[0] * (1.0f / 256.0f);
    y2[base + tid] = (_Float16)(d * rsqrtf(var + 1e-5f) * w[tid] + b[tid]);
}

// ---------------- QKV GEMM: xr[100352,256] @ Wqkv[768,256]^T + b, scattered ----------------
__global__ __launch_bounds__(256) void k_qkv(const _Float16* __restrict__ A,
                                             const _Float16* __restrict__ W,
                                             const float* __restrict__ bias,
                                             _Float16* __restrict__ q,
                                             _Float16* __restrict__ k,
                                             _Float16* __restrict__ vT) {
    const int lane = threadIdx.x & 31, wv = threadIdx.x >> 5;
    const int job = blockIdx.x * 8 + wv;          // 6272*12 jobs
    const int mt = job / 12, nb = job - mt * 12;
    const int m0 = mt * 16, n0 = nb * 64;
    const int lr = lane & 15, mh = (lane & 16) ? 8 : 0;
    v8f z = {}; v8f c[4] = { z, z, z, z };
    gemm16x64(A, W, m0, n0, lane, c);
#pragma unroll
    for (int j = 0; j < 4; ++j) {
        const int n = n0 + j * 16 + lr;
        const int sec = n >> 8, ci = n & 255;
        const int head = ci >> 5, dh = ci & 31;
        const float bv = bias[n];
#pragma unroll
        for (int r = 0; r < 8; ++r) {
            const int t = m0 + r + mh;
            const int win = t / 49, tw = t - win * 49;
            const size_t wh = (size_t)win * N_HEAD + head;
            const _Float16 val = (_Float16)(c[j][r] + bv);
            if (sec == 0)      q[(wh * N_TOK + tw) * D_HEAD + dh] = val;
            else if (sec == 1) k[(wh * N_TOK + tw) * D_HEAD + dh] = val;
            else               vT[(wh * D_HEAD + dh) * 64 + tw]   = val;
        }
    }
}

// ---------------- fused shifted-window attention (1 block = 1 window, wave = head) ----------------
DEV int regionOf(int p) {
    const int r = p / 56, c = p - r * 56;
    const int rr = (r < 49) ? 0 : ((r < 53) ? 1 : 2);
    const int cc = (c < 49) ? 0 : ((c < 53) ? 1 : 2);
    return rr * 3 + cc;
}

__global__ __launch_bounds__(256) void k_attn(const _Float16* __restrict__ q,
                                              const _Float16* __restrict__ kb,
                                              const _Float16* __restrict__ vT,
                                              _Float16* __restrict__ obuf) {
    __shared__ __align__(32) _Float16 asb[8][64 * 64];   // 64 KB, per-head prob tiles
    const int lane = threadIdx.x & 31, head = threadIdx.x >> 5;
    const int g = blockIdx.x;                 // global window
    const int w = g & 63;                     // mask window id
    const size_t wh = (size_t)g * N_HEAD + head;
    const _Float16* Q = q  + wh * N_TOK * D_HEAD;
    const _Float16* K = kb + wh * N_TOK * D_HEAD;
    const _Float16* V = vT + wh * D_HEAD * 64;
    const int lr = lane & 15;
    const int hiA = (lane & 16) ? 8 : 0;      // accumulator row offset (rows 8..15)
    const int hiB = (lane & 16) ? 16 : 0;     // B-frag K sub-offset

    // ---- load Q (A-frags) and K^T (B-frags), zero-padded past 49 ----
    v16h aq[4], bk[4];
    const v16h z16 = {};
    const v8h  z8  = {};
#pragma unroll
    for (int t = 0; t < 4; ++t) {
        const int m = t * 16 + lr;
        union { v16h v; v8h h[2]; } u;
        if (m < N_TOK) {
            u.v = *(const v16h*)(Q + m * D_HEAD + hiB);
        } else { u.h[0] = z8; u.h[1] = z8; }
        aq[t] = u.v;
        const int n = t * 16 + lr;
        bk[t] = (n < N_TOK) ? *(const v16h*)(K + n * D_HEAD + hiB) : z16;
    }

    // ---- scores = Q K^T  (K-dim = DH = 32 -> one WMMA per tile) ----
    v8f s[4][4];
#pragma unroll
    for (int mt = 0; mt < 4; ++mt)
#pragma unroll
        for (int nt = 0; nt < 4; ++nt) {
            v8f z = {};
            s[mt][nt] = __builtin_amdgcn_wmma_f32_16x16x32_f16(
                false, aq[mt], false, bk[nt], (short)0, z, false, false);
        }

    // ---- analytic shift-mask + row softmax (rows live across 16-lane halves) ----
    const float scale = 0.17677669529663687f;   // 32^-0.5
    int regn[4]; bool nok[4];
#pragma unroll
    for (int nt = 0; nt < 4; ++nt) {
        const int n = nt * 16 + lr;
        nok[nt] = (n < N_TOK);
        regn[nt] = regionOf(w * N_TOK + n);
    }
#pragma unroll
    for (int mt = 0; mt < 4; ++mt) {
#pragma unroll
        for (int r = 0; r < 8; ++r) {
            const int m = mt * 16 + r + hiA;   // hiA==8 exactly when lane>=16
            const int rm = regionOf(w * N_TOK + m);
            float vals[4]; float vmax = -3.0e38f;
#pragma unroll
            for (int nt = 0; nt < 4; ++nt) {
                float v = s[mt][nt][r] * scale + ((regn[nt] == rm) ? 0.0f : -100.0f);
                if (!nok[nt]) v = -3.0e38f;
                vals[nt] = v; vmax = fmaxf(vmax, v);
            }
#pragma unroll
            for (int d = 1; d < 16; d <<= 1) vmax = fmaxf(vmax, __shfl_xor(vmax, d, 32));
            float sum = 0.0f;
#pragma unroll
            for (int nt = 0; nt < 4; ++nt) {
                float e = (vals[nt] <= -1.0e37f) ? 0.0f : __expf(vals[nt] - vmax);
                vals[nt] = e; sum += e;
            }
#pragma unroll
            for (int d = 1; d < 16; d <<= 1) sum += __shfl_xor(sum, d, 32);
            const float inv = (sum > 0.0f) ? (1.0f / sum) : 0.0f;
            const bool mok = (m < N_TOK);
#pragma unroll
            for (int nt = 0; nt < 4; ++nt)
                asb[head][m * 64 + nt * 16 + lr] =
                    (_Float16)((mok && nok[nt]) ? vals[nt] * inv : 0.0f);
        }
    }
    __syncthreads();

    // ---- O = P @ V   (K padded to 64 -> two 16x16x32 steps) ----
    v8f o[4][2];
#pragma unroll
    for (int mt = 0; mt < 4; ++mt) { v8f z = {}; o[mt][0] = z; o[mt][1] = z; }
#pragma unroll
    for (int ks = 0; ks < 2; ++ks) {
        v16h aa[4], bb[2];
#pragma unroll
        for (int mt = 0; mt < 4; ++mt) {
            const int m = mt * 16 + lr;
            const _Float16* p = &asb[head][m * 64 + ks * 32 + hiB];
            union { v16h v; v8h h[2]; } u;
            u.v = *(const v16h*)p;
            aa[mt] = u.v;
        }
#pragma unroll
        for (int nt = 0; nt < 2; ++nt) {
            const int n = nt * 16 + lr;                      // dh
            bb[nt] = *(const v16h*)(V + n * 64 + ks * 32 + hiB);
        }
#pragma unroll
        for (int mt = 0; mt < 4; ++mt)
#pragma unroll
            for (int nt = 0; nt < 2; ++nt)
                o[mt][nt] = __builtin_amdgcn_wmma_f32_16x16x32_f16(
                    false, aa[mt], false, bb[nt], (short)0, o[mt][nt], false, false);
    }

    // ---- store merged-head output in window-token order ----
    const size_t tokbase = (size_t)g * N_TOK;
#pragma unroll
    for (int mt = 0; mt < 4; ++mt)
#pragma unroll
        for (int r = 0; r < 8; ++r) {
            const int m = mt * 16 + r + hiA;
            if (m < N_TOK) {
#pragma unroll
                for (int nt = 0; nt < 2; ++nt) {
                    const int dh = nt * 16 + lr;
                    obuf[(tokbase + m) * C_DIM + head * D_HEAD + dh] =
                        (_Float16)o[mt][nt][r];
                }
            }
        }
}

// ---------------- proj GEMM + window_reverse + roll(-3,-3) + residual ----------------
__global__ __launch_bounds__(256) void k_proj(const _Float16* __restrict__ A,
                                              const _Float16* __restrict__ W,
                                              const float* __restrict__ bias,
                                              const float* __restrict__ xin,
                                              float* __restrict__ x1) {
    const int lane = threadIdx.x & 31, wv = threadIdx.x >> 5;
    const int job = blockIdx.x * 8 + wv;          // 6272*4 jobs
    const int mt = job >> 2, nb = job & 3;
    const int m0 = mt * 16, n0 = nb * 64;
    const int lr = lane & 15, mh = (lane & 16) ? 8 : 0;
    v8f z = {}; v8f c[4] = { z, z, z, z };
    gemm16x64(A, W, m0, n0, lane, c);
#pragma unroll
    for (int j = 0; j < 4; ++j) {
        const int n = n0 + j * 16 + lr;
        const float bv = bias[n];
#pragma unroll
        for (int r = 0; r < 8; ++r) {
            const int t = m0 + r + mh;
            const int bi = t / 3136, p = t - bi * 3136;
            const int win = p / 49, tw = p - win * 49;
            const int h0 = win >> 3, w0 = win & 7;
            const int hw = tw / 7,  ww = tw - hw * 7;
            int rr = h0 * 7 + hw + 53; if (rr >= 56) rr -= 56;   // reverse + roll(-3)
            int cc = w0 * 7 + ww + 53; if (cc >= 56) cc -= 56;
            const size_t idx = (((size_t)bi * 3136) + rr * 56 + cc) * C_DIM + n;
            x1[idx] = xin[idx] + c[j][r] + bv;
        }
    }
}

// ---------------- fc1 GEMM + exact GELU ----------------
__global__ __launch_bounds__(256) void k_fc1(const _Float16* __restrict__ A,
                                             const _Float16* __restrict__ W,
                                             const float* __restrict__ bias,
                                             _Float16* __restrict__ hid) {
    const int lane = threadIdx.x & 31, wv = threadIdx.x >> 5;
    const int job = blockIdx.x * 8 + wv;
    const int mt = job >> 2, nb = job & 3;
    const int m0 = mt * 16, n0 = nb * 64;
    const int lr = lane & 15, mh = (lane & 16) ? 8 : 0;
    v8f z = {}; v8f c[4] = { z, z, z, z };
    gemm16x64(A, W, m0, n0, lane, c);
#pragma unroll
    for (int j = 0; j < 4; ++j) {
        const int n = n0 + j * 16 + lr;
        const float bv = bias[n];
#pragma unroll
        for (int r = 0; r < 8; ++r) {
            const int t = m0 + r + mh;
            const float v = c[j][r] + bv;
            const float g = 0.5f * v * (1.0f + erff(v * 0.7071067811865475f));
            hid[(size_t)t * C_DIM + n] = (_Float16)g;
        }
    }
}

// ---------------- fc2 GEMM + residual -> final output ----------------
__global__ __launch_bounds__(256) void k_fc2(const _Float16* __restrict__ A,
                                             const _Float16* __restrict__ W,
                                             const float* __restrict__ bias,
                                             const float* __restrict__ x1,
                                             float* __restrict__ out) {
    const int lane = threadIdx.x & 31, wv = threadIdx.x >> 5;
    const int job = blockIdx.x * 8 + wv;
    const int mt = job >> 2, nb = job & 3;
    const int m0 = mt * 16, n0 = nb * 64;
    const int lr = lane & 15, mh = (lane & 16) ? 8 : 0;
    v8f z = {}; v8f c[4] = { z, z, z, z };
    gemm16x64(A, W, m0, n0, lane, c);
#pragma unroll
    for (int j = 0; j < 4; ++j) {
        const int n = n0 + j * 16 + lr;
        const float bv = bias[n];
#pragma unroll
        for (int r = 0; r < 8; ++r) {
            const size_t idx = (size_t)(m0 + r + mh) * C_DIM + n;
            out[idx] = x1[idx] + c[j][r] + bv;
        }
    }
}

// ---------------- host-side launch sequence ----------------
extern "C" void kernel_launch(void* const* d_in, const int* in_sizes, int n_in,
                              void* d_out, int out_size, void* d_ws, size_t ws_size,
                              hipStream_t stream) {
    (void)in_sizes; (void)n_in; (void)out_size; (void)ws_size;
    const float* x      = (const float*)d_in[0];
    const float* ln1_w  = (const float*)d_in[1];
    const float* ln1_b  = (const float*)d_in[2];
    const float* qkv_w  = (const float*)d_in[3];
    const float* qkv_b  = (const float*)d_in[4];
    const float* proj_w = (const float*)d_in[5];
    const float* proj_b = (const float*)d_in[6];
    const float* ln2_w  = (const float*)d_in[7];
    const float* ln2_b  = (const float*)d_in[8];
    const float* fc1_w  = (const float*)d_in[9];
    const float* fc1_b  = (const float*)d_in[10];
    const float* fc2_w  = (const float*)d_in[11];
    const float* fc2_b  = (const float*)d_in[12];
    float* out = (float*)d_out;

    char* ws = (char*)d_ws;
    _Float16* xr   = (_Float16*)(ws + OFF_XR);   // also LN2 output
    _Float16* qb   = (_Float16*)(ws + OFF_Q);    // also fc1 hidden
    _Float16* kb   = (_Float16*)(ws + OFF_K);
    _Float16* vT   = (_Float16*)(ws + OFF_VT);
    _Float16* ob   = (_Float16*)(ws + OFF_O);
    float*    x1   = (float*)   (ws + OFF_X1);
    _Float16* wqkv = (_Float16*)(ws + OFF_WQKV);
    _Float16* wprj = (_Float16*)(ws + OFF_WPRJ);
    _Float16* wf1  = (_Float16*)(ws + OFF_WFC1);
    _Float16* wf2  = (_Float16*)(ws + OFF_WFC2);

    // zero the K-padded V^T buffer (pads must be 0 for the padded WMMA K-loop)
    k_zero32<<<8192, 256, 0, stream>>>((uint32_t*)vT, SZ_VT16 / 4);

    // weights -> f16
    k_cvt16<<<(768 * 256 + 255) / 256, 256, 0, stream>>>(qkv_w, wqkv, 768 * 256);
    k_cvt16<<<(256 * 256 + 255) / 256, 256, 0, stream>>>(proj_w, wprj, 256 * 256);
    k_cvt16<<<(256 * 256 + 255) / 256, 256, 0, stream>>>(fc1_w, wf1, 256 * 256);
    k_cvt16<<<(256 * 256 + 255) / 256, 256, 0, stream>>>(fc2_w, wf2, 256 * 256);

    // LN1 + roll + window-order gather
    k_ln1<<<TOK, 256, 0, stream>>>(x, ln1_w, ln1_b, xr);

    // QKV projection (WMMA), scattered to per-head Q/K and K-padded V^T
    k_qkv<<<(MTILES * 12) / 8, 256, 0, stream>>>(xr, wqkv, qkv_b, qb, kb, vT);

    // fused masked attention (WMMA QK^T -> softmax -> WMMA PV)
    k_attn<<<NWIN, 256, 0, stream>>>(qb, kb, vT, ob);

    // proj + window_reverse + roll + residual
    k_proj<<<(MTILES * 4) / 8, 256, 0, stream>>>(ob, wprj, proj_b, x, x1);

    // LN2 (reuse xr region as y2)
    k_ln2<<<TOK, 256, 0, stream>>>(x1, ln2_w, ln2_b, xr);

    // MLP: fc1+GELU (hidden reuses Q region), fc2+residual -> out
    k_fc1<<<(MTILES * 4) / 8, 256, 0, stream>>>(xr, wf1, fc1_b, qb);
    k_fc2<<<(MTILES * 4) / 8, 256, 0, stream>>>(qb, wf2, fc2_b, x1, out);
}
